// StructuredTemporalAttentionLayer_62019327754878
// MI455X (gfx1250) — matrix operand order repaired and checked by reading, not compiled, not run on hardware
//
#include <hip/hip_runtime.h>

typedef __attribute__((ext_vector_type(16))) __bf16 v16bf;
typedef __attribute__((ext_vector_type(8)))  float  v8f;

#define KDIM 512
#define HDIM 256

__device__ __forceinline__ unsigned f2bf_pack(float lo, float hi) {
  union { float f; unsigned u; } a, b;
  a.f = lo; b.f = hi;
  unsigned ra = (a.u + 0x7FFFu + ((a.u >> 16) & 1u)) >> 16;
  unsigned rb = (b.u + 0x7FFFu + ((b.u >> 16) & 1u)) >> 16;
  return (ra & 0xFFFFu) | (rb << 16);
}

// Repack W1 (KDIM x HDIM fp32, row-major) into bf16 WMMA-B fragments:
// dword index = ((kt*16 + nt)*32 + lane)*8 + v
// lane 0-15: N = nt*16+lane, K rows {kt*32 + 2v, +1}; lanes 16-31: K += 16.
__global__ void k_pack_w1(const float* __restrict__ w1, unsigned* __restrict__ w1p) {
  int tid  = blockIdx.x * blockDim.x + threadIdx.x;   // 0..65535
  int v    = tid & 7;
  int lane = (tid >> 3) & 31;
  int nt   = (tid >> 8) & 15;
  int kt   = tid >> 12;
  int n = nt * 16 + (lane & 15);
  int k = kt * 32 + ((lane >> 4) ? 16 : 0) + 2 * v;
  w1p[tid] = f2bf_pack(w1[k * HDIM + n], w1[(k + 1) * HDIM + n]);
}

// Fused unary FF: X(rows x 512) @ W1 -> ReLU -> @ w2(256x2) + b2 -> unary(rows x 2)
// Block = 256 threads (8 waves); block tile = 64 rows x 256 cols.
__global__ __launch_bounds__(256)
void k_unary(const float* __restrict__ X, const unsigned* __restrict__ W1p,
             const float* __restrict__ b1, const float* __restrict__ w2,
             const float* __restrict__ b2, float* __restrict__ unary,
             int rowsTotal) {
  __shared__ __align__(16) unsigned short xs[64 * 48];  // 64 rows, pitch 48 bf16 (96B)
  __shared__ float up[4][16][2];

  const int tid  = threadIdx.x;
  const int lane = tid & 31;
  const int w    = tid >> 5;
  const int rt   = w >> 1;            // row tile 0..3 (16 rows each)
  const int ntBase = (w & 1) * 8;     // 8 n-tiles of 16 cols
  const int blockRow = blockIdx.x * 64;

  if (tid < 128) ((float*)up)[tid] = 0.f;

  v8f acc[8] = {};

  // X -> LDS loader assignment: each thread handles 8 K elements of one row
  const int lrow = tid >> 2;
  const int lkg  = (tid & 3) * 8;
  long grow = blockRow + lrow;
  if (grow >= rowsTotal) grow = rowsTotal - 1;
  const float* xrow = X + (size_t)grow * KDIM;

  // A-fragment source in LDS (16-bit A 16x32 layout)
  const int am    = rt * 16 + (lane & 15);
  const int akoff = (lane >> 4) * 16;
  const unsigned short* ap = &xs[am * 48 + akoff];

  for (int kt = 0; kt < KDIM / 32; ++kt) {
    __syncthreads();
    {
      const float4* xp = (const float4*)(xrow + kt * 32 + lkg);
      float4 xa = xp[0];
      float4 xb = xp[1];
      uint4 pk;
      pk.x = f2bf_pack(xa.x, xa.y);
      pk.y = f2bf_pack(xa.z, xa.w);
      pk.z = f2bf_pack(xb.x, xb.y);
      pk.w = f2bf_pack(xb.z, xb.w);
      *(uint4*)&xs[lrow * 48 + lkg] = pk;
      if (kt + 1 < KDIM / 32)
        __builtin_prefetch(xrow + (kt + 1) * 32 + lkg, 0, 1);
    }
    __syncthreads();

    union { uint4 u[2]; v16bf v; } A;
    A.u[0] = *(const uint4*)ap;
    A.u[1] = *(const uint4*)(ap + 8);  // +16 bytes: K = koff+8..koff+15

#pragma unroll
    for (int t = 0; t < 8; ++t) {
      const uint4* bp =
          (const uint4*)(W1p + ((size_t)((kt * 16 + ntBase + t) * 32 + lane) * 8));
      union { uint4 u[2]; v16bf v; } Bf;
      Bf.u[0] = bp[0];
      Bf.u[1] = bp[1];
      acc[t] = __builtin_amdgcn_wmma_f32_16x16x32_bf16(
          false, A.v, false, Bf.v, (short)0, acc[t], false, false);
    }
  }

  // Epilogue: bias + ReLU + layer2 (256 -> 2)
  float p0[8], p1[8];
#pragma unroll
  for (int v = 0; v < 8; ++v) { p0[v] = 0.f; p1[v] = 0.f; }
#pragma unroll
  for (int t = 0; t < 8; ++t) {
    int n = (ntBase + t) * 16 + (lane & 15);
    float bb  = b1[n];
    float w20 = w2[n * 2 + 0];
    float w21 = w2[n * 2 + 1];
#pragma unroll
    for (int v = 0; v < 8; ++v) {
      float h = acc[t][v] + bb;
      h = fmaxf(h, 0.f);
      p0[v] += h * w20;
      p1[v] += h * w21;
    }
  }
  // reduce over the 16 lanes (distinct n) in each half-wave
#pragma unroll
  for (int v = 0; v < 8; ++v) {
#pragma unroll
    for (int off = 8; off > 0; off >>= 1) {
      p0[v] += __shfl_xor(p0[v], off, 16);
      p1[v] += __shfl_xor(p1[v], off, 16);
    }
  }
  if ((lane & 15) == 0) {
    int mofs = (lane >> 4) * 8;   // lane 0 -> rows v, lane 16 -> rows v+8
#pragma unroll
    for (int v = 0; v < 8; ++v) {
      atomicAdd(&up[rt][mofs + v][0], p0[v]);  // exactly 2 adds/cell -> deterministic
      atomicAdd(&up[rt][mofs + v][1], p1[v]);
    }
  }
  __syncthreads();
  if (tid < 128) {
    int rt2 = tid >> 5;
    int m   = (tid >> 1) & 15;
    int o   = tid & 1;
    long g = blockRow + rt2 * 16 + m;
    if (g < rowsTotal) unary[g * 2 + o] = up[rt2][m][o] + b2[o];
  }
}

// Relation encoder: ctx (rows x 5) -> FF 5->64->3 -> 2x2 table [r0, r1, r1, r2]
__global__ void k_rel(const float* __restrict__ ctx, const float* __restrict__ rw1,
                      const float* __restrict__ rb1, const float* __restrict__ rw2,
                      const float* __restrict__ rb2, float* __restrict__ binary,
                      int nrows) {
  int r = blockIdx.x * blockDim.x + threadIdx.x;
  if (r >= nrows) return;
  float x0 = ctx[r * 5 + 0], x1 = ctx[r * 5 + 1], x2 = ctx[r * 5 + 2];
  float x3 = ctx[r * 5 + 3], x4 = ctx[r * 5 + 4];
  float a0 = rb2[0], a1 = rb2[1], a2 = rb2[2];
#pragma unroll 4
  for (int j = 0; j < 64; ++j) {
    float h = rb1[j] + x0 * rw1[j] + x1 * rw1[64 + j] + x2 * rw1[128 + j] +
              x3 * rw1[192 + j] + x4 * rw1[256 + j];
    h = fmaxf(h, 0.f);
    a0 += h * rw2[j * 3 + 0];
    a1 += h * rw2[j * 3 + 1];
    a2 += h * rw2[j * 3 + 2];
  }
  float4 o; o.x = a0; o.y = a1; o.z = a1; o.w = a2;
  *(float4*)(binary + (size_t)r * 4) = o;
}

// Factor graph: one wave per NB row; 128 configs across lanes (4/lane).
__global__ __launch_bounds__(256)
void k_joint(const float* __restrict__ unary, const float* __restrict__ binary,
             float* __restrict__ out, int NB, int Bsz) {
  __shared__ float tab[8][100];
  int lane = threadIdx.x & 31;
  int w = threadIdx.x >> 5;
  int row = blockIdx.x * 8 + w;
  if (row >= NB) row = NB - 1;  // clamp: duplicate rows rewrite identical values
  int b = row % Bsz;
  const float* u  = unary  + (size_t)row * 14;
  const float* bb = binary + (size_t)b * 84;
  for (int idx = lane; idx < 98; idx += 32)
    tab[w][idx] = (idx < 14) ? u[idx] : bb[idx - 14];
  __syncthreads();
  const float* tu = tab[w];
  const float* tb = tab[w] + 14;

  constexpr int PI[21] = {0,0,0,0,0,0,1,1,1,1,1,2,2,2,2,3,3,3,4,4,5};
  constexpr int PJ[21] = {1,2,3,4,5,6,2,3,4,5,6,3,4,5,6,4,5,6,5,6,6};

  float jv[4];
#pragma unroll
  for (int q = 0; q < 4; ++q) {
    int c = q * 32 + lane;
    int bit[7];
    float s = 0.f;
#pragma unroll
    for (int i = 0; i < 7; ++i) {
      bit[i] = (c >> (6 - i)) & 1;  // reference C-order reshape: entity 0 = MSB
      s += tu[i * 2 + bit[i]];
    }
#pragma unroll
    for (int p = 0; p < 21; ++p)
      s += tb[p * 4 + bit[PI[p]] * 2 + bit[PJ[p]]];
    jv[q] = s;
  }

  float m = fmaxf(fmaxf(jv[0], jv[1]), fmaxf(jv[2], jv[3]));
#pragma unroll
  for (int off = 16; off > 0; off >>= 1) m = fmaxf(m, __shfl_xor(m, off, 32));
  float sum = 0.f;
#pragma unroll
  for (int q = 0; q < 4; ++q) sum += expf(jv[q] - m);
#pragma unroll
  for (int off = 16; off > 0; off >>= 1) sum += __shfl_xor(sum, off, 32);
  float logZ = m + logf(sum);

  float* orow = out + (size_t)row * 142;
  float acc14[14];
#pragma unroll
  for (int k = 0; k < 14; ++k) acc14[k] = 0.f;
#pragma unroll
  for (int q = 0; q < 4; ++q) {
    int c = q * 32 + lane;
    float jn = jv[q] - logZ;
    orow[q * 32 + lane] = jn;
    float e = expf(jn);
#pragma unroll
    for (int i = 0; i < 7; ++i) {
      int bi = (c >> (6 - i)) & 1;
      acc14[i * 2 + 0] += bi ? 0.f : e;   // constant indices -> stays in VGPRs
      acc14[i * 2 + 1] += bi ? e : 0.f;
    }
  }
#pragma unroll
  for (int k = 0; k < 14; ++k) {
#pragma unroll
    for (int off = 16; off > 0; off >>= 1)
      acc14[k] += __shfl_xor(acc14[k], off, 32);
  }
  if (lane < 14) orow[128 + lane] = logf(acc14[lane]);
}

extern "C" void kernel_launch(void* const* d_in, const int* in_sizes, int n_in,
                              void* d_out, int out_size, void* d_ws, size_t ws_size,
                              hipStream_t stream) {
  (void)n_in; (void)out_size; (void)ws_size;
  const float* inp = (const float*)d_in[0];
  const float* ctx = (const float*)d_in[1];
  const float* uw1 = (const float*)d_in[2];
  const float* ub1 = (const float*)d_in[3];
  const float* uw2 = (const float*)d_in[4];
  const float* ub2 = (const float*)d_in[5];
  const float* rw1 = (const float*)d_in[6];
  const float* rb1 = (const float*)d_in[7];
  const float* rw2 = (const float*)d_in[8];
  const float* rb2 = (const float*)d_in[9];
  float* out = (float*)d_out;

  int Bsz = in_sizes[1] / (21 * 5);                  // 1024
  int N   = in_sizes[0] / (Bsz * 7 * KDIM);          // 12
  int NB  = N * Bsz;                                 // 12288
  int rowsTotal = NB * 7;                            // 86016

  char* ws = (char*)d_ws;
  unsigned* w1p   = (unsigned*)ws;                   // 256 KB packed bf16 W1
  size_t uOff = 262144;
  float* unaryBuf = (float*)(ws + uOff);             // rowsTotal*2 floats
  size_t uBytes = ((size_t)rowsTotal * 2 * sizeof(float) + 255) & ~(size_t)255;
  float* binBuf   = (float*)(ws + uOff + uBytes);    // Bsz*21*4 floats

  k_pack_w1<<<(KDIM / 32) * (HDIM / 16), 256, 0, stream>>>(uw1, w1p);  // 256 blocks

  int relRows = Bsz * 21;
  k_rel<<<(relRows + 255) / 256, 256, 0, stream>>>(ctx, rw1, rb1, rw2, rb2, binBuf,
                                                   relRows);

  k_unary<<<(rowsTotal + 63) / 64, 256, 0, stream>>>(inp, w1p, ub1, uw2, ub2,
                                                     unaryBuf, rowsTotal);

  k_joint<<<(NB + 7) / 8, 256, 0, stream>>>(unaryBuf, binBuf, out, NB, Bsz);
}
